// PaddedAttentionTT_70781061038450
// MI455X (gfx1250) — matrix operand order repaired and checked by reading, not compiled, not run on hardware
//
#include <hip/hip_runtime.h>
#include <hip/hip_bf16.h>

typedef __bf16 bf16;
typedef __attribute__((ext_vector_type(16))) __bf16 v16bf;
typedef __attribute__((ext_vector_type(8)))  float  v8f;
typedef __attribute__((ext_vector_type(4)))  __bf16 v4bf;

#define B_  2
#define S_  2048
#define H_  4096
#define NH_ 32
#define HD_ 128

// ---------------------------------------------------------------------------
// CDNA5 async global->LDS copy (ASYNCcnt path), inline asm (portable across
// ROCm 7.2 / amdgpu-toolchain per bridge doc). LDS address = low 32 bits of
// the flat address (LDS aperture keeps the offset in addr[31:0]).
// ---------------------------------------------------------------------------
__device__ inline void async_ld_b128(void* lds, const void* g) {
  unsigned l = (unsigned)(uintptr_t)lds;
  unsigned long long a = (unsigned long long)(uintptr_t)g;
  asm volatile("global_load_async_to_lds_b128 %0, %1, off"
               :: "v"(l), "v"(a) : "memory");
}
__device__ inline void wait_async0() {
  asm volatile("s_wait_asynccnt 0x0" ::: "memory");
}

// ---------------------------------------------------------------------------
// WMMA helpers (CDNA5 wave32 layouts)
// A 16x32 bf16 : lane L holds row M=L&15; elems 0..7 -> K = (L>=16?8:0)+0..7,
//                elems 8..15 -> +16.  Contiguous 16B + 16B reads.
// B 32x16 bf16 : lane L holds col N=L&15; elems 0..7 -> K = (L>=16?8:0)+0..7,
//                elems 8..15 -> +16.  Contiguous 16B + 16B reads.
// C/D 16x16 f32: vgpr r, lane L -> M = r + 8*(L>>4), N = L&15.
// ---------------------------------------------------------------------------
__device__ inline v16bf load_a_frag(const bf16* p0, int ld) {
  const int lane = threadIdx.x & 31;
  const bf16* p = p0 + (size_t)(lane & 15) * ld + ((lane >> 4) << 3);
  union { v16bf v; uint4 q[2]; } u;
  u.q[0] = *(const uint4*)(p);
  u.q[1] = *(const uint4*)(p + 16);
  return u.v;
}

__device__ inline v16bf load_b_frag(const bf16* p0, int ld) {
  const int lane = threadIdx.x & 31;
  const bf16* p = p0 + (size_t)(lane & 15) * ld + ((lane >> 4) << 3);
  union { v16bf v; uint4 q[2]; } u;
  u.q[0] = *(const uint4*)(p);
  u.q[1] = *(const uint4*)(p + 16);
  return u.v;
}

__device__ inline v8f wmma_bf16(v16bf a, v16bf b, v8f c) {
  return __builtin_amdgcn_wmma_f32_16x16x32_bf16(false, a, false, b,
                                                 (short)0, c, false, false);
}

// ---------------------------------------------------------------------------
// fp32 -> bf16 conversion
// ---------------------------------------------------------------------------
__global__ void cvt_f32_bf16(const float4* __restrict__ in,
                             bf16* __restrict__ out, int n4) {
  int i = blockIdx.x * blockDim.x + threadIdx.x;
  if (i >= n4) return;
  float4 v = in[i];
  v4bf o;
  o.x = (bf16)v.x; o.y = (bf16)v.y; o.z = (bf16)v.z; o.w = (bf16)v.w;
  *(v4bf*)(out + (size_t)i * 4) = o;
}

// ---------------------------------------------------------------------------
// C = A(MxK, bf16 row-major) * W(NxK, bf16 row-major)^T
// MODE 0: bf16 [M,N]   MODE 1: f32 [M,N]   MODE 2: bf16 V-transposed [B][N][S]
// Block tile 128x128, 8 waves of 32x64, K-step 32.
// Double-buffered LDS fed by async global->LDS b128 copies (ASYNCcnt).
// ---------------------------------------------------------------------------
template<int MODE>
__global__ __launch_bounds__(256)
void gemm_bt(const bf16* __restrict__ A, const bf16* __restrict__ W,
             void* __restrict__ Cp, int M, int N, int K) {
  __shared__ bf16 As[2][128][32];   // 16 KB
  __shared__ bf16 Bs[2][128][32];   // 16 KB  (Bs[n][k])
  const int m0   = blockIdx.y * 128;
  const int n0   = blockIdx.x * 128;
  const int w    = threadIdx.x >> 5;
  const int lane = threadIdx.x & 31;
  const int wm   = w >> 1;   // 0..3 -> 32-row slab
  const int wn   = w & 1;    // 0..1 -> 64-col slab

  // Per-thread slice of the cooperative tile copy: 2 x b128 per matrix.
  const int trow = threadIdx.x >> 1;          // 0..127
  const int tc0  = (threadIdx.x & 1) * 16;    // 0 or 16 (elements)

  auto load_tile = [&](int buf, int k0) {
    const bf16* ga = A + (size_t)(m0 + trow) * K + k0 + tc0;
    const bf16* gb = W + (size_t)(n0 + trow) * K + k0 + tc0;
    async_ld_b128(&As[buf][trow][tc0],     ga);
    async_ld_b128(&As[buf][trow][tc0 + 8], ga + 8);
    async_ld_b128(&Bs[buf][trow][tc0],     gb);
    async_ld_b128(&Bs[buf][trow][tc0 + 8], gb + 8);
  };

  v8f acc[2][4] = {};

  load_tile(0, 0);
  wait_async0();
  __syncthreads();

  int cur = 0;
  for (int k0 = 0; k0 < K; k0 += 32) {
    if (k0 + 32 < K) load_tile(cur ^ 1, k0 + 32);   // async into other buffer
    if (k0 + 64 < K) {                               // L2 warm one tile further
      __builtin_prefetch(A + (size_t)(m0 + trow) * K + k0 + 64, 0, 1);
      __builtin_prefetch(W + (size_t)(n0 + trow) * K + k0 + 64, 0, 1);
    }

    v16bf af[2];
#pragma unroll
    for (int mi = 0; mi < 2; ++mi)
      af[mi] = load_a_frag(&As[cur][wm * 32 + mi * 16][0], 32);
#pragma unroll
    for (int ni = 0; ni < 4; ++ni) {
      v16bf bfg = load_b_frag(&Bs[cur][wn * 64 + ni * 16][0], 32);
#pragma unroll
      for (int mi = 0; mi < 2; ++mi)
        acc[mi][ni] = wmma_bf16(af[mi], bfg, acc[mi][ni]);
    }

    wait_async0();
    __syncthreads();
    cur ^= 1;
  }

#pragma unroll
  for (int mi = 0; mi < 2; ++mi)
#pragma unroll
    for (int ni = 0; ni < 4; ++ni)
#pragma unroll
      for (int r = 0; r < 8; ++r) {
        int m = m0 + wm * 32 + mi * 16 + r + ((lane >> 4) << 3);
        int n = n0 + wn * 64 + ni * 16 + (lane & 15);
        float v = acc[mi][ni][r];
        if (MODE == 0) {
          ((bf16*)Cp)[(size_t)m * N + n] = (bf16)v;
        } else if (MODE == 1) {
          ((float*)Cp)[(size_t)m * N + n] = v;
        } else {  // V transposed: [b][d][s]
          int bb = m >> 11;            // m / S_
          int s  = m & (S_ - 1);
          ((bf16*)Cp)[((size_t)bb * N + n) * S_ + s] = (bf16)v;
        }
      }
}

// ---------------------------------------------------------------------------
// RoPE (in place, bf16). One thread owns the (d, d+64) pair -> safe in place.
// ---------------------------------------------------------------------------
__global__ void rope_q(bf16* __restrict__ Q, const float* __restrict__ cs,
                       const float* __restrict__ sn) {
  size_t tid = (size_t)blockIdx.x * blockDim.x + threadIdx.x;
  if (tid >= (size_t)B_ * S_ * NH_ * 64) return;
  int d   = tid & 63;
  int h   = (tid >> 6) & (NH_ - 1);
  int row = (int)(tid >> 11);          // b*S + s
  int s   = row & (S_ - 1);
  bf16* p = Q + (size_t)row * H_ + h * HD_;
  float x1 = (float)p[d], x2 = (float)p[d + 64];
  float c1 = cs[s * HD_ + d],      s1 = sn[s * HD_ + d];
  float c2 = cs[s * HD_ + d + 64], s2 = sn[s * HD_ + d + 64];
  p[d]      = (bf16)(x1 * c1 - x2 * s1);
  p[d + 64] = (bf16)(x2 * c2 + x1 * s2);
}

__global__ void rope_k(bf16* __restrict__ Kk, const float* __restrict__ cs,
                       const float* __restrict__ sn) {
  size_t tid = (size_t)blockIdx.x * blockDim.x + threadIdx.x;
  if (tid >= (size_t)B_ * S_ * 64) return;
  int d   = tid & 63;
  int row = (int)(tid >> 6);           // b*S + s
  int s   = row & (S_ - 1);
  bf16* p = Kk + (size_t)row * HD_;
  float x1 = (float)p[d], x2 = (float)p[d + 64];
  float c1 = cs[s * HD_ + d],      s1 = sn[s * HD_ + d];
  float c2 = cs[s * HD_ + d + 64], s2 = sn[s * HD_ + d + 64];
  p[d]      = (bf16)(x1 * c1 - x2 * s1);
  p[d + 64] = (bf16)(x2 * c2 + x1 * s2);
}

// ---------------------------------------------------------------------------
// Flash attention fwd. One wave per 16-query tile; 32-key steps with causal
// early exit. Q: [b*S, H] per-head slices (RoPE'd), K: [b*S, HD] (RoPE'd),
// V: [B][HD][S] transposed. Online softmax through per-wave LDS tile.
// K/V fragments preloaded into registers so the load clause overlaps WMMAs.
// ---------------------------------------------------------------------------
__global__ __launch_bounds__(128)
void attn_fwd(const bf16* __restrict__ Q, const bf16* __restrict__ Kt,
              const bf16* __restrict__ Vt, bf16* __restrict__ O) {
  __shared__ float sc[4][16][32];     // raw scores
  __shared__ bf16  pt[4][16][32];     // probabilities (A-layout staging)
  __shared__ float mrow[4][16], lrow[4][16], arow[4][16];

  const int w    = threadIdx.x >> 5;
  const int lane = threadIdx.x & 31;
  const int q0   = (blockIdx.x * 4 + w) * 16;
  const int h    = blockIdx.y;
  const int b    = blockIdx.z;
  const float scale = 0.08838834764831845f;   // 1/sqrt(128)

  // Q fragments (4 x 16x32 bf16)
  v16bf qa[4];
  const bf16* qbase = Q + ((size_t)(b * S_ + q0)) * H_ + h * HD_;
#pragma unroll
  for (int c = 0; c < 4; ++c) qa[c] = load_a_frag(qbase + c * 32, H_);

  v8f acc[8] = {};
  if (lane < 16) { mrow[w][lane] = -1e30f; lrow[w][lane] = 0.0f; }

  const int kmax   = q0 + 15;
  const int ntiles = (q0 + 16 + 31) >> 5;   // 32-key tiles, causal bound

  for (int kt = 0; kt < ntiles; ++kt) {
    const int K0 = kt * 32;
#pragma unroll
    for (int sub = 0; sub < 2; ++sub) {
      const int kb = K0 + sub * 16;
      if (kb <= kmax) {
        const bf16* kbase = Kt + ((size_t)(b * S_ + kb)) * HD_;
        v16bf kf[4];
#pragma unroll
        for (int c = 0; c < 4; ++c)          // one load clause, then 4 WMMAs
          kf[c] = load_b_frag(kbase + c * 32, HD_);
        v8f s = {};
#pragma unroll
        for (int c = 0; c < 4; ++c) s = wmma_bf16(qa[c], kf[c], s);
#pragma unroll
        for (int r = 0; r < 8; ++r) {
          int m = r + ((lane >> 4) << 3);
          int n = lane & 15;
          float v = s[r] * scale;
          if (kb + n > q0 + m) v = -1e30f;    // causal mask
          sc[w][m][sub * 16 + n] = v;
        }
      } else {
#pragma unroll
        for (int r = 0; r < 8; ++r)
          sc[w][r + ((lane >> 4) << 3)][sub * 16 + (lane & 15)] = -1e30f;
      }
    }

    // online softmax update: one lane per row (intra-wave LDS, in order)
    if (lane < 16) {
      float mold = mrow[w][lane];
      float tmax = -1e30f;
#pragma unroll
      for (int j = 0; j < 32; ++j) tmax = fmaxf(tmax, sc[w][lane][j]);
      float mnew  = fmaxf(mold, tmax);
      float alpha = __expf(mold - mnew);
      float psum  = 0.0f;
#pragma unroll
      for (int j = 0; j < 32; ++j) {
        float p = __expf(sc[w][lane][j] - mnew);
        pt[w][lane][j] = (bf16)p;
        psum += p;
      }
      mrow[w][lane] = mnew;
      lrow[w][lane] = lrow[w][lane] * alpha + psum;
      arow[w][lane] = alpha;
    }

    // rescale accumulators by per-row alpha
    float al[8];
#pragma unroll
    for (int r = 0; r < 8; ++r) al[r] = arow[w][r + ((lane >> 4) << 3)];
#pragma unroll
    for (int nt = 0; nt < 8; ++nt)
#pragma unroll
      for (int r = 0; r < 8; ++r) acc[nt][r] *= al[r];

    // P(16x32) x V(32x128): preload 8 B fragments, then 8 WMMAs
    v16bf pf = load_a_frag(&pt[w][0][0], 32);
    const bf16* vbase = Vt + (size_t)b * HD_ * S_ + K0;
    v16bf vf[8];
#pragma unroll
    for (int nt = 0; nt < 8; ++nt)
      vf[nt] = load_b_frag(vbase + (size_t)(nt * 16) * S_, S_);
#pragma unroll
    for (int nt = 0; nt < 8; ++nt)
      acc[nt] = wmma_bf16(pf, vf[nt], acc[nt]);
  }

  // normalize and store
  if (lane < 16) arow[w][lane] = 1.0f / lrow[w][lane];
  float il[8];
#pragma unroll
  for (int r = 0; r < 8; ++r) il[r] = arow[w][r + ((lane >> 4) << 3)];
  bf16* obase = O + ((size_t)(b * S_ + q0)) * H_ + h * HD_;
#pragma unroll
  for (int nt = 0; nt < 8; ++nt)
#pragma unroll
    for (int r = 0; r < 8; ++r) {
      int m = r + ((lane >> 4) << 3);
      int n = nt * 16 + (lane & 15);
      obase[(size_t)m * H_ + n] = (bf16)(acc[nt][r] * il[r]);
    }
}

// ---------------------------------------------------------------------------
// Host orchestration
// ---------------------------------------------------------------------------
extern "C" void kernel_launch(void* const* d_in, const int* in_sizes, int n_in,
                              void* d_out, int out_size, void* d_ws, size_t ws_size,
                              hipStream_t stream) {
  (void)in_sizes; (void)n_in; (void)out_size; (void)ws_size;
  const float* hs = (const float*)d_in[0];
  const float* cs = (const float*)d_in[1];
  const float* sn = (const float*)d_in[2];
  const float* Wq = (const float*)d_in[3];
  const float* Wk = (const float*)d_in[4];
  const float* Wv = (const float*)d_in[5];
  const float* Wd = (const float*)d_in[6];

  char*  ws  = (char*)d_ws;
  size_t off = 0;
  auto alloc = [&](size_t elems) -> bf16* {
    bf16* p = (bf16*)(ws + off);
    off += ((elems * sizeof(bf16) + 255) & ~(size_t)255);
    return p;
  };
  bf16* hsb = alloc((size_t)B_ * S_ * H_);   // hidden bf16
  bf16* wqb = alloc((size_t)H_ * H_);
  bf16* wkb = alloc((size_t)HD_ * H_);
  bf16* wvb = alloc((size_t)HD_ * H_);
  bf16* wdb = alloc((size_t)H_ * H_);
  bf16* Qt  = alloc((size_t)B_ * S_ * H_);   // Q (RoPE'd in place)
  bf16* Kt  = alloc((size_t)B_ * S_ * HD_);  // K (RoPE'd in place)
  bf16* Vt  = alloc((size_t)B_ * HD_ * S_);  // V transposed [b][d][s]
  bf16* Ob  = alloc((size_t)B_ * S_ * H_);   // attention output

  auto cvt = [&](const float* in, bf16* out, size_t n) {
    int n4 = (int)(n / 4);
    cvt_f32_bf16<<<(n4 + 255) / 256, 256, 0, stream>>>((const float4*)in, out, n4);
  };
  cvt(hs, hsb, (size_t)B_ * S_ * H_);
  cvt(Wq, wqb, (size_t)H_ * H_);
  cvt(Wk, wkb, (size_t)HD_ * H_);
  cvt(Wv, wvb, (size_t)HD_ * H_);
  cvt(Wd, wdb, (size_t)H_ * H_);

  // projections
  gemm_bt<0><<<dim3(H_ / 128, (B_ * S_) / 128), 256, 0, stream>>>(
      hsb, wqb, Qt, B_ * S_, H_, H_);
  gemm_bt<0><<<dim3(1, (B_ * S_) / 128), 256, 0, stream>>>(
      hsb, wkb, Kt, B_ * S_, HD_, H_);
  gemm_bt<2><<<dim3(1, (B_ * S_) / 128), 256, 0, stream>>>(
      hsb, wvb, Vt, B_ * S_, HD_, H_);

  // RoPE
  {
    size_t nq = (size_t)B_ * S_ * NH_ * 64;
    rope_q<<<(unsigned)((nq + 255) / 256), 256, 0, stream>>>(Qt, cs, sn);
    size_t nk = (size_t)B_ * S_ * 64;
    rope_k<<<(unsigned)((nk + 255) / 256), 256, 0, stream>>>(Kt, cs, sn);
  }

  // flash attention
  attn_fwd<<<dim3(S_ / 64, NH_, B_), 128, 0, stream>>>(Qt, Kt, Vt, Ob);

  // output projection (fp32 out)
  gemm_bt<1><<<dim3(H_ / 128, (B_ * S_) / 128), 256, 0, stream>>>(
      Ob, wdb, (float*)d_out, B_ * S_, H_, H_);
}
